// ConditionalMolDecoder_5987184410767
// MI455X (gfx1250) — matrix operand order repaired and checked by reading, not compiled, not run on hardware
//
#include <hip/hip_runtime.h>

// ---------------------------------------------------------------------------
// Conditional GRU molecular decoder for MI455X (gfx1250).
//
// Roofline: ~4.1 TFLOP of GEMM over 119 sequential steps; weights (8.5 MB in
// bf16) are L2-resident (192 MB L2), HBM traffic is only the 250 MB logits
// output -> compute/L2-bound. Use v_wmma_f32_16x16x32_bf16 (bf16 in, fp32
// accumulate). Batch rows are fully independent (argmax feedback is per-row),
// so a persistent workgroup owns 32 rows for all 119 steps: weight L2 traffic
// = 128 WGs x 119 steps x 8.5 MB ~= 130 GB.
// ---------------------------------------------------------------------------

#define VOCAB   128
#define CCOND   3
#define EMBD    128
#define HID     512
#define LAT     256
#define NLAY    3
#define NSTEP   119          // T-1
#define BATCH   4096
#define H3      (3*HID)      // 1536
#define KZP     288          // Z+C=259 padded to 32-multiple
#define KX0P    160          // E+C=131 padded to 32-multiple
#define RPW     32           // batch rows per workgroup (two 16-row WMMA tiles)
#define NWG     (BATCH/RPW)  // 128 workgroups
#define NTHR    128          // 4 waves of 32

typedef __bf16        v16bf __attribute__((ext_vector_type(16)));
typedef float         v8f   __attribute__((ext_vector_type(8)));
typedef unsigned int  v4u   __attribute__((ext_vector_type(4)));

union AF { v16bf v; v4u q[2]; };

__device__ __forceinline__ unsigned short f2bf(float f) {
  unsigned int u = __float_as_uint(f);
  u += 0x7FFFu + ((u >> 16) & 1u);          // round-to-nearest-even
  return (unsigned short)(u >> 16);
}
__device__ __forceinline__ float bf2f(unsigned short h) {
  return __uint_as_float(((unsigned int)h) << 16);
}
__device__ __forceinline__ float sigf(float x) { return 1.0f / (1.0f + __expf(-x)); }
__device__ __forceinline__ float tanhfast(float x) {
  x = fminf(15.0f, fmaxf(-15.0f, x));
  float e = __expf(2.0f * x);
  return (e - 1.0f) / (e + 1.0f);
}

// A fragment (16x32 bf16, row-major source): lanes 0-15 carry K {kb+0..7, kb+16..23},
// lanes 16-31 carry K {kb+8..15, kb+24..31} (CDNA5 ISA 7.12.2 16-bit A layout).
__device__ __forceinline__ v16bf lds_a_frag(const unsigned short* base, int row,
                                            int stride, int kb, int hi) {
  const unsigned short* p = base + (size_t)row * stride + kb + (hi ? 8 : 0);
  AF f;
  f.q[0] = *(const v4u*)p;
  f.q[1] = *(const v4u*)(p + 16);
  return f.v;
}
// B fragment (32x16 bf16): lane n holds K {kb+0..7, kb+16..23} (lanes 0-15)
// or {kb+8..15, kb+24..31} (lanes 16-31) of weight row n  (matches A layout).
__device__ __forceinline__ v16bf glb_b_frag(const unsigned short* wrow, int kb, int hi) {
  const unsigned short* p = wrow + kb + (hi ? 8 : 0);
  AF f;
  f.q[0] = *(const v4u*)p;
  f.q[1] = *(const v4u*)(p + 16);
  return f.v;
}
__device__ __forceinline__ v8f wmma_bf(v16bf a, v16bf b, v8f c) {
  return __builtin_amdgcn_wmma_f32_16x16x32_bf16(false, a, false, b, (short)0, c,
                                                 false, false);
}

// ---------------------------------------------------------------------------
// fp32 -> bf16 conversion with row K-padding (weights / embedding prep).
// ---------------------------------------------------------------------------
__global__ void cvt_pad_bf16(const float* __restrict__ src, unsigned short* __restrict__ dst,
                             int rows, int ks, int kd) {
  int i = blockIdx.x * blockDim.x + threadIdx.x;
  if (i >= rows * kd) return;
  int r = i / kd, k = i - r * kd;
  float v = (k < ks) ? src[(size_t)r * ks + k] : 0.0f;
  dst[i] = f2bf(v);
}

// ---------------------------------------------------------------------------
// Persistent decoder: one WG = 32 batch rows x 119 steps.
// LDS: hbf[3][32][512] bf16 (98304 B) | stage[32][512] bf16 (32768 B) |
//      ubuf 18432 B (union: zc[32][288]bf16 / x[32][160]bf16 / logits[32][128]f32) |
//      condbf[32][3] bf16 | tok[32] int   => 149824 B total.
// ---------------------------------------------------------------------------
#define HBF_B    (NLAY*RPW*HID*2)
#define STAGE_B  (RPW*HID*2)
#define UBUF_B   18432
#define SMEM_B   (HBF_B + STAGE_B + UBUF_B + 192 + 128)

__global__ void __launch_bounds__(NTHR, 1)
gru_decoder_persistent(const float* __restrict__ z,
                       const float* __restrict__ cond,
                       const unsigned short* __restrict__ emb_bf,
                       const unsigned short* __restrict__ wz_bf,
                       const float* __restrict__ b_z,
                       const unsigned short* __restrict__ wih0_bf,
                       const unsigned short* __restrict__ wih1_bf,
                       const unsigned short* __restrict__ wih2_bf,
                       const unsigned short* __restrict__ whh_bf,
                       const float* __restrict__ b_ih,
                       const float* __restrict__ b_hh,
                       const unsigned short* __restrict__ wout_bf,
                       const float* __restrict__ b_out,
                       float* __restrict__ out) {
  extern __shared__ __align__(16) char smem[];
  unsigned short* hbf    = (unsigned short*)smem;               // [3][32][512]
  unsigned short* stage  = hbf + (size_t)NLAY * RPW * HID;      // [32][512]
  char*           ubuf   = (char*)(stage + (size_t)RPW * HID);  // union buffer
  unsigned short* condbf = (unsigned short*)(ubuf + UBUF_B);    // [32][3]
  int*            tok    = (int*)(condbf + 96);                 // [32]

  const int tid    = threadIdx.x;
  const int wave   = tid >> 5;
  const int lane   = tid & 31;
  const int lane15 = lane & 15;
  const int hi     = lane >> 4;
  const int rowbase = blockIdx.x * RPW;
  const int wcol    = wave * 128;          // this wave's 128-col slice of H

  const v8f Z8 = {};

  // ---- init: zc (bf16, padded), cond bf16, start tokens ----
  unsigned short* zc = (unsigned short*)ubuf;                   // [32][288]
  for (int i = tid; i < RPW * KZP; i += NTHR) {
    int m = i / KZP, k = i - m * KZP;
    float v = 0.0f;
    if (k < LAT)              v = z[(size_t)(rowbase + m) * LAT + k];
    else if (k < LAT + CCOND) v = cond[(size_t)(rowbase + m) * CCOND + (k - LAT)];
    zc[i] = f2bf(v);
  }
  if (tid < RPW * CCOND) {
    int m = tid / CCOND, c = tid - m * CCOND;
    condbf[tid] = f2bf(cond[(size_t)(rowbase + m) * CCOND + c]);
  }
  if (tid < RPW) tok[tid] = 1;             // start token = 1
  __syncthreads();

  // ---- h0 = tanh(zc @ w_z^T + b_z), sections s=0..2 map to layers ----
  for (int jt = 0; jt < 8; ++jt) {
    int colg = wcol + jt * 16;
    v8f acc[3][2];
#pragma unroll
    for (int s = 0; s < 3; ++s) { acc[s][0] = Z8; acc[s][1] = Z8; }
    for (int kb = 0; kb < KZP; kb += 32) {
      v16bf a0 = lds_a_frag(zc, lane15, KZP, kb, hi);
      v16bf a1 = lds_a_frag(zc, 16 + lane15, KZP, kb, hi);
#pragma unroll
      for (int s = 0; s < 3; ++s) {
        const unsigned short* wr = wz_bf + (size_t)(s * HID + colg + lane15) * KZP;
        v16bf bf = glb_b_frag(wr, kb, hi);
        acc[s][0] = wmma_bf(a0, bf, acc[s][0]);
        acc[s][1] = wmma_bf(a1, bf, acc[s][1]);
      }
    }
    int jc = colg + lane15;
#pragma unroll
    for (int s = 0; s < 3; ++s) {
      float bz = b_z[s * HID + jc];
#pragma unroll
      for (int rt = 0; rt < 2; ++rt)
#pragma unroll
        for (int v = 0; v < 8; ++v) {
          int m = rt * 16 + hi * 8 + v;
          hbf[(size_t)s * RPW * HID + m * HID + jc] = f2bf(tanhfast(acc[s][rt][v] + bz));
        }
    }
  }
  __syncthreads();

  unsigned short* xbuf = (unsigned short*)ubuf;   // [32][160] bf16
  float*          lg   = (float*)ubuf;            // [32][128] f32 logits

  // ======================= autoregressive steps =======================
  for (int step = 0; step < NSTEP; ++step) {
    // x = [emb[tok], cond, 0-pad] in bf16
    for (int i = tid; i < RPW * KX0P; i += NTHR) {
      int m = i / KX0P, k = i - m * KX0P;
      unsigned short v;
      if (k < EMBD)              v = emb_bf[tok[m] * EMBD + k];
      else if (k < EMBD + CCOND) v = condbf[m * CCOND + (k - EMBD)];
      else                       v = 0;
      xbuf[i] = v;
    }
    __syncthreads();

    for (int l = 0; l < NLAY; ++l) {
      const unsigned short* xsrc = (l == 0) ? xbuf : (hbf + (size_t)(l - 1) * RPW * HID);
      const int             kx   = (l == 0) ? KX0P : HID;
      const unsigned short* wih  = (l == 0) ? wih0_bf : ((l == 1) ? wih1_bf : wih2_bf);
      const unsigned short* whh  = whh_bf + (size_t)l * H3 * HID;
      const unsigned short* hsrc = hbf + (size_t)l * RPW * HID;

      for (int jt = 0; jt < 8; ++jt) {
        int colg = wcol + jt * 16;
        v8f ai[3][2], ah[3][2];
#pragma unroll
        for (int g = 0; g < 3; ++g) { ai[g][0] = Z8; ai[g][1] = Z8; ah[g][0] = Z8; ah[g][1] = Z8; }
        // gi = x @ w_ih^T  (three gate column blocks share the same j slice)
        for (int kb = 0; kb < kx; kb += 32) {
          v16bf a0 = lds_a_frag(xsrc, lane15, kx, kb, hi);
          v16bf a1 = lds_a_frag(xsrc, 16 + lane15, kx, kb, hi);
#pragma unroll
          for (int g = 0; g < 3; ++g) {
            const unsigned short* wr = wih + (size_t)(g * HID + colg + lane15) * kx;
            v16bf bf = glb_b_frag(wr, kb, hi);
            ai[g][0] = wmma_bf(a0, bf, ai[g][0]);
            ai[g][1] = wmma_bf(a1, bf, ai[g][1]);
          }
        }
        // gh = h_old @ w_hh^T
        for (int kb = 0; kb < HID; kb += 32) {
          v16bf a0 = lds_a_frag(hsrc, lane15, HID, kb, hi);
          v16bf a1 = lds_a_frag(hsrc, 16 + lane15, HID, kb, hi);
#pragma unroll
          for (int g = 0; g < 3; ++g) {
            const unsigned short* wr = whh + (size_t)(g * HID + colg + lane15) * HID;
            v16bf bf = glb_b_frag(wr, kb, hi);
            ah[g][0] = wmma_bf(a0, bf, ah[g][0]);
            ah[g][1] = wmma_bf(a1, bf, ah[g][1]);
          }
        }
        // GRU gate math (fp32) -> staged new h (bf16)
        int jc = colg + lane15;
        float bir = b_ih[l * H3 + jc];
        float biz = b_ih[l * H3 + HID + jc];
        float bin = b_ih[l * H3 + 2 * HID + jc];
        float bhr = b_hh[l * H3 + jc];
        float bhz = b_hh[l * H3 + HID + jc];
        float bhn = b_hh[l * H3 + 2 * HID + jc];
#pragma unroll
        for (int rt = 0; rt < 2; ++rt)
#pragma unroll
          for (int v = 0; v < 8; ++v) {
            int m = rt * 16 + hi * 8 + v;
            float hold = bf2f(hsrc[m * HID + jc]);
            float r = sigf(ai[0][rt][v] + bir + ah[0][rt][v] + bhr);
            float u = sigf(ai[1][rt][v] + biz + ah[1][rt][v] + bhz);
            float n = tanhfast(ai[2][rt][v] + bin + r * (ah[2][rt][v] + bhn));
            stage[m * HID + jc] = f2bf((1.0f - u) * n + u * hold);
          }
      }
      __syncthreads();   // all gh reads of hbf[l] are done
      {                   // commit own 128-col slice: stage -> hbf[l]
        unsigned short* hdst = hbf + (size_t)l * RPW * HID;
        for (int i = lane; i < RPW * 16; i += 32) {
          int m = i >> 4;
          int c = (i & 15) * 8;
          *(v4u*)&hdst[m * HID + wcol + c] = *(const v4u*)&stage[m * HID + wcol + c];
        }
      }
      __syncthreads();   // hbf[l] now holds new h, usable as next layer's x
    }

    // logits = h2 @ w_out^T + b_out ; each wave does a 32-col slice of V
    {
      const unsigned short* hsrc = hbf + (size_t)2 * RPW * HID;
      v8f lacc[2][2];
      lacc[0][0] = Z8; lacc[0][1] = Z8; lacc[1][0] = Z8; lacc[1][1] = Z8;
      int vb = wave * 32;
      for (int kb = 0; kb < HID; kb += 32) {
        v16bf a0 = lds_a_frag(hsrc, lane15, HID, kb, hi);
        v16bf a1 = lds_a_frag(hsrc, 16 + lane15, HID, kb, hi);
#pragma unroll
        for (int ct = 0; ct < 2; ++ct) {
          const unsigned short* wr = wout_bf + (size_t)(vb + ct * 16 + lane15) * HID;
          v16bf bf = glb_b_frag(wr, kb, hi);
          lacc[ct][0] = wmma_bf(a0, bf, lacc[ct][0]);
          lacc[ct][1] = wmma_bf(a1, bf, lacc[ct][1]);
        }
      }
#pragma unroll
      for (int ct = 0; ct < 2; ++ct) {
        int nc = vb + ct * 16 + lane15;
        float bo = b_out[nc];
#pragma unroll
        for (int rt = 0; rt < 2; ++rt)
#pragma unroll
          for (int v = 0; v < 8; ++v) {
            int m = rt * 16 + hi * 8 + v;
            lg[m * VOCAB + nc] = lacc[ct][rt][v] + bo;
          }
      }
    }
    __syncthreads();

    // coalesced logits write-out: out[b][step][v]
    {
      size_t obase = (size_t)rowbase * NSTEP * VOCAB + (size_t)step * VOCAB;
      for (int i = tid; i < RPW * VOCAB; i += NTHR) {
        int m = i >> 7, v = i & 127;
        out[obase + (size_t)m * NSTEP * VOCAB + v] = lg[i];
      }
    }
    // per-row argmax (first occurrence of max), wave 0 only
    if (tid < RPW) {
      const float* row = lg + tid * VOCAB;
      float best = row[0]; int bi = 0;
      for (int v = 1; v < VOCAB; ++v) { float x = row[v]; if (x > best) { best = x; bi = v; } }
      tok[tid] = bi;
    }
    __syncthreads();   // protect tok + lg/xbuf reuse
  }
}

// ---------------------------------------------------------------------------
extern "C" void kernel_launch(void* const* d_in, const int* in_sizes, int n_in,
                              void* d_out, int out_size, void* d_ws, size_t ws_size,
                              hipStream_t stream) {
  const float* z        = (const float*)d_in[0];
  const float* cond     = (const float*)d_in[1];
  const float* emb      = (const float*)d_in[2];
  const float* w_z      = (const float*)d_in[3];
  const float* b_z      = (const float*)d_in[4];
  const float* w_ih0    = (const float*)d_in[5];
  const float* w_ih_r   = (const float*)d_in[6];
  const float* w_hh     = (const float*)d_in[7];
  const float* b_ih     = (const float*)d_in[8];
  const float* b_hh     = (const float*)d_in[9];
  const float* w_out    = (const float*)d_in[10];
  const float* b_out    = (const float*)d_in[11];
  float* out = (float*)d_out;

  char* ws = (char*)d_ws;
  size_t off = 0;
  auto alloc = [&](size_t bytes) {
    void* p = ws + off;
    off = (off + bytes + 255) & ~(size_t)255;
    return p;
  };
  unsigned short* wz_bf   = (unsigned short*)alloc((size_t)H3 * KZP * 2);
  unsigned short* wih0_bf = (unsigned short*)alloc((size_t)H3 * KX0P * 2);
  unsigned short* wih1_bf = (unsigned short*)alloc((size_t)H3 * HID * 2);
  unsigned short* wih2_bf = (unsigned short*)alloc((size_t)H3 * HID * 2);
  unsigned short* whh_bf  = (unsigned short*)alloc((size_t)3 * H3 * HID * 2);
  unsigned short* wout_bf = (unsigned short*)alloc((size_t)VOCAB * HID * 2);
  unsigned short* emb_bf  = (unsigned short*)alloc((size_t)VOCAB * EMBD * 2);

  auto cvt = [&](const float* s, unsigned short* d, int rows, int ks, int kd) {
    int n = rows * kd;
    cvt_pad_bf16<<<(n + 255) / 256, 256, 0, stream>>>(s, d, rows, ks, kd);
  };
  cvt(w_z,                  wz_bf,   H3, LAT + CCOND, KZP);   // 259 -> 288
  cvt(w_ih0,                wih0_bf, H3, EMBD + CCOND, KX0P); // 131 -> 160
  cvt(w_ih_r,               wih1_bf, H3, HID, HID);
  cvt(w_ih_r + (size_t)H3 * HID, wih2_bf, H3, HID, HID);
  cvt(w_hh,                 whh_bf,  3 * H3, HID, HID);
  cvt(w_out,                wout_bf, VOCAB, HID, HID);
  cvt(emb,                  emb_bf,  VOCAB, EMBD, EMBD);

  gru_decoder_persistent<<<NWG, NTHR, SMEM_B, stream>>>(
      z, cond, emb_bf, wz_bf, b_z, wih0_bf, wih1_bf, wih2_bf, whh_bf,
      b_ih, b_hh, wout_bf, b_out, out);
}
